// ProAffinityGNN_68487548502616
// MI455X (gfx1250) — matrix operand order repaired and checked by reading, not compiled, not run on hardware
//
#include <hip/hip_runtime.h>
#include <hip/hip_bf16.h>
#include <math.h>

// Problem constants (match reference)
#define N_NODES   50000
#define N_EDGES   800000
#define N_GRAPHS  64
#define DIN       128
#define H         64
#define H3        192      // 3*H
#define H2        128      // 2*H
#define T_STEPS   3
#define NEG_SLOPE 0.2f

typedef __attribute__((ext_vector_type(16))) _Float16 v16h;
typedef __attribute__((ext_vector_type(8)))  float    v8f;

__device__ __forceinline__ v8f wmma_f16(v16h a, v16h b, v8f c) {
  // D = A(16x32 f16) * B(32x16 f16) + C(16x16 f32)
  return __builtin_amdgcn_wmma_f32_16x16x32_f16(
      /*neg_a=*/false, a, /*neg_b=*/false, b,
      /*c_mod=*/(short)0, c, /*reuse_a=*/false, /*reuse_b=*/false);
}

// A fragment: 16x32 tile (row-major f32 in memory, converted to f16).
// ISA layout: lanes 0-15 hold M=lane, K={kb..kb+7, kb+16..kb+23}, kb=0;
// lanes 16-31 same M=lane&15 with kb=8.
__device__ __forceinline__ v16h load_a_f32(const float* __restrict__ base,
                                           int ld, int lane) {
  const int m  = lane & 15;
  const int kb = (lane < 16) ? 0 : 8;
  const float* row = base + m * ld;
  v16h a;
#pragma unroll
  for (int i = 0; i < 8; ++i) a[i] = (_Float16)row[kb + i];
#pragma unroll
  for (int i = 0; i < 8; ++i) a[8 + i] = (_Float16)row[kb + 16 + i];
  return a;
}

// B fragment: 32x16 tile (K x N, row-major f32, leading dim ld).
// Lanes 0-15: N=lane, K={0..7,16..23} ; lanes 16-31: N=lane&15, K={8..15,24..31}.
__device__ __forceinline__ v16h load_b_f32(const float* __restrict__ base,
                                           int ld, int lane) {
  const int n  = lane & 15;
  const int kb = (lane < 16) ? 0 : 8;
  v16h b;
#pragma unroll
  for (int i = 0; i < 16; ++i) b[i] = (_Float16)base[(kb + (i & 7) + ((i >> 3) << 4)) * ld + n];
  return b;
}

__device__ __forceinline__ float sigmoidf(float x) {
  return 1.0f / (1.0f + __expf(-x));
}

// Float atomic max via int/uint punning (valid with -inf initialization).
__device__ __forceinline__ void atomicMaxFloat(float* addr, float val) {
  if (val >= 0.0f) atomicMax((int*)addr, __float_as_int(val));
  else             atomicMin((unsigned int*)addr, __float_as_uint(val));
}

// ---------------------------------------------------------------------------
// h = relu(x @ W_in + b_in)   [N,128]x[128,64] -> [N,64]
// grid.x = N/16 ; block = 128 (4 waves, wave w owns output-col tile w)
// ---------------------------------------------------------------------------
__global__ void k_input_proj(const float* __restrict__ x,
                             const float* __restrict__ W,
                             const float* __restrict__ b,
                             float* __restrict__ h) {
  const int wave  = threadIdx.x >> 5;
  const int lane  = threadIdx.x & 31;
  const int row0  = blockIdx.x * 16;
  const int ncol0 = wave * 16;

  v8f acc = {};
#pragma unroll
  for (int kc = 0; kc < DIN / 32; ++kc) {
    v16h a  = load_a_f32(x + row0 * DIN + kc * 32, DIN, lane);
    v16h bb = load_b_f32(W + kc * 32 * H + ncol0, H, lane);
    acc = wmma_f16(a, bb, acc);
  }
  const int n  = lane & 15;
  const int mb = (lane < 16) ? 0 : 8;
  const float bias = b[ncol0 + n];
#pragma unroll
  for (int r = 0; r < 8; ++r) {
    float v = acc[r] + bias;
    v = v > 0.0f ? v : 0.0f;
    h[(row0 + mb + r) * H + ncol0 + n] = v;
  }
}

// ---------------------------------------------------------------------------
// Per-timestep scratch init: m=0, segmax=-inf, segsum=0
// ---------------------------------------------------------------------------
__global__ void k_init_step(float* __restrict__ m, float* __restrict__ segmax,
                            float* __restrict__ segsum) {
  const int i = blockIdx.x * blockDim.x + threadIdx.x;
  if (i < N_NODES * H) m[i] = 0.0f;
  if (i < N_NODES) { segmax[i] = -INFINITY; segsum[i] = 0.0f; }
}

// ---------------------------------------------------------------------------
// Per-node attention scalars: s_src = h . a[:H], s_dst = h . a[H:]
// ---------------------------------------------------------------------------
__global__ void k_att_scalar(const float* __restrict__ h,
                             const float* __restrict__ a_att,
                             float* __restrict__ s_src,
                             float* __restrict__ s_dst) {
  const int i = blockIdx.x * blockDim.x + threadIdx.x;
  if (i >= N_NODES) return;
  const float* row = h + i * H;
  float ss = 0.0f, sd = 0.0f;
#pragma unroll 8
  for (int j = 0; j < H; ++j) {
    const float v = row[j];
    ss += v * a_att[j];
    sd += v * a_att[H + j];
  }
  s_src[i] = ss;
  s_dst[i] = sd;
}

// ---------------------------------------------------------------------------
// Edge scores + segment max (leaky_relu(s_src[src]+s_dst[dst]))
// ---------------------------------------------------------------------------
__global__ void k_edge_score(const int* __restrict__ src,
                             const int* __restrict__ dst,
                             const float* __restrict__ s_src,
                             const float* __restrict__ s_dst,
                             float* __restrict__ score,
                             float* __restrict__ segmax) {
  const int e = blockIdx.x * blockDim.x + threadIdx.x;
  if (e >= N_EDGES) return;
  const int d = dst[e];
  float sc = s_src[src[e]] + s_dst[d];
  sc = sc > 0.0f ? sc : NEG_SLOPE * sc;
  score[e] = sc;
  atomicMaxFloat(&segmax[d], sc);
}

// ---------------------------------------------------------------------------
// Edge exp + segment sum (score buffer overwritten with exp values)
// ---------------------------------------------------------------------------
__global__ void k_edge_exp(const int* __restrict__ dst,
                           float* __restrict__ score,
                           const float* __restrict__ segmax,
                           float* __restrict__ segsum) {
  const int e = blockIdx.x * blockDim.x + threadIdx.x;
  if (e >= N_EDGES) return;
  const int d = dst[e];
  const float ex = __expf(score[e] - segmax[d]);
  score[e] = ex;
  atomicAdd(&segsum[d], ex);
}

// ---------------------------------------------------------------------------
// Weighted message scatter: m[dst] += alpha * h[src].  One wave per edge;
// lanes cover the 64 hidden dims in two coalesced steps.
// ---------------------------------------------------------------------------
__global__ void k_edge_msg(const int* __restrict__ src,
                           const int* __restrict__ dst,
                           const float* __restrict__ ex,
                           const float* __restrict__ segsum,
                           const float* __restrict__ h,
                           float* __restrict__ m) {
  const int wid  = (blockIdx.x * blockDim.x + threadIdx.x) >> 5;
  const int lane = threadIdx.x & 31;
  if (wid >= N_EDGES) return;
  const int s = src[wid];
  const int d = dst[wid];
  const float alpha = ex[wid] / (segsum[d] + 1e-16f);
  atomicAdd(&m[d * H + lane],      alpha * h[s * H + lane]);
  atomicAdd(&m[d * H + lane + 32], alpha * h[s * H + lane + 32]);
}

// ---------------------------------------------------------------------------
// Fused GRU cell: h = GRU(m, h).  One wave handles 16 nodes x all 192 gate
// columns.  r/z gates: single accumulator holds m@Wi + h@Wh + bi + bh.
// n gate: gi_n and gh_n kept separate for gi_n + r*gh_n.
// block = 128 (4 waves, one 16-node tile each)
// ---------------------------------------------------------------------------
__global__ void k_gru(const float* __restrict__ mbuf,
                      float* __restrict__ h,
                      const float* __restrict__ Wi,
                      const float* __restrict__ Wh,
                      const float* __restrict__ bi,
                      const float* __restrict__ bh) {
  const int wave = threadIdx.x >> 5;
  const int lane = threadIdx.x & 31;
  const int tile = blockIdx.x * 4 + wave;
  if (tile >= N_NODES / 16) return;
  const int row0 = tile * 16;
  const int n    = lane & 15;
  const int mb   = (lane < 16) ? 0 : 8;

  // A fragments (K = 64 -> two 32-wide chunks), f32 -> f16 in-register.
  const v16h mA0 = load_a_f32(mbuf + row0 * H + 0,  H, lane);
  const v16h mA1 = load_a_f32(mbuf + row0 * H + 32, H, lane);
  const v16h hA0 = load_a_f32(h    + row0 * H + 0,  H, lane);
  const v16h hA1 = load_a_f32(h    + row0 * H + 32, H, lane);

  // r and z gates: columns [0, 128), 8 tiles, fused gi+gh accumulation.
  v8f accRZ[8];
#pragma unroll
  for (int t = 0; t < 8; ++t) {
    const int c0 = t * 16;
    const float bsum = bi[c0 + n] + bh[c0 + n];
    v8f acc;
#pragma unroll
    for (int r = 0; r < 8; ++r) acc[r] = bsum;
    acc = wmma_f16(hA0, load_b_f32(Wh + c0,            H3, lane), acc);
    acc = wmma_f16(hA1, load_b_f32(Wh + 32 * H3 + c0,  H3, lane), acc);
    acc = wmma_f16(mA0, load_b_f32(Wi + c0,            H3, lane), acc);
    acc = wmma_f16(mA1, load_b_f32(Wi + 32 * H3 + c0,  H3, lane), acc);
    accRZ[t] = acc;
  }

  // n gate: columns [128, 192), gi and gh kept separate.
  v8f gin[4], ghn[4];
#pragma unroll
  for (int t = 0; t < 4; ++t) {
    const int c0 = H2 + t * 16;
    v8f ai, ah;
    const float bvi = bi[c0 + n];
    const float bvh = bh[c0 + n];
#pragma unroll
    for (int r = 0; r < 8; ++r) { ai[r] = bvi; ah[r] = bvh; }
    ai = wmma_f16(mA0, load_b_f32(Wi + c0,           H3, lane), ai);
    ai = wmma_f16(mA1, load_b_f32(Wi + 32 * H3 + c0, H3, lane), ai);
    ah = wmma_f16(hA0, load_b_f32(Wh + c0,           H3, lane), ah);
    ah = wmma_f16(hA1, load_b_f32(Wh + 32 * H3 + c0, H3, lane), ah);
    gin[t] = ai;
    ghn[t] = ah;
  }

  // Elementwise GRU update directly on C-fragment registers.
  // Fragment slot (lane, r) of tile t maps to node row0+mb+r, hidden j=t*16+n,
  // and the same slot in accRZ[t] / accRZ[t+4] / gin[t] / ghn[t] is the
  // r-gate / z-gate / gi_n / gh_n value for that (node, j).
#pragma unroll
  for (int t = 0; t < 4; ++t) {
#pragma unroll
    for (int r = 0; r < 8; ++r) {
      const int node = row0 + mb + r;
      const float rg = sigmoidf(accRZ[t][r]);
      const float zg = sigmoidf(accRZ[t + 4][r]);
      const float ng = tanhf(gin[t][r] + rg * ghn[t][r]);
      const float ho = h[node * H + t * 16 + n];
      h[node * H + t * 16 + n] = (1.0f - zg) * ng + zg * ho;
    }
  }
}

// ---------------------------------------------------------------------------
// Pooling init + scatter
// ---------------------------------------------------------------------------
__global__ void k_init_pool(float* __restrict__ psum, float* __restrict__ pmax,
                            float* __restrict__ pcnt) {
  const int i = blockIdx.x * blockDim.x + threadIdx.x;
  if (i < N_GRAPHS * H) { psum[i] = 0.0f; pmax[i] = -INFINITY; }
  if (i < N_GRAPHS) pcnt[i] = 0.0f;
}

__global__ void k_pool(const float* __restrict__ h,
                       const int* __restrict__ batch,
                       float* __restrict__ psum, float* __restrict__ pmax,
                       float* __restrict__ pcnt) {
  const int wid  = (blockIdx.x * blockDim.x + threadIdx.x) >> 5;
  const int lane = threadIdx.x & 31;
  if (wid >= N_NODES) return;
  const int g = batch[wid];
  const float v0 = h[wid * H + lane];
  const float v1 = h[wid * H + lane + 32];
  atomicAdd(&psum[g * H + lane], v0);
  atomicAdd(&psum[g * H + lane + 32], v1);
  atomicMaxFloat(&pmax[g * H + lane], v0);
  atomicMaxFloat(&pmax[g * H + lane + 32], v1);
  if (lane == 0) atomicAdd(&pcnt[g], 1.0f);
}

// ---------------------------------------------------------------------------
// Head: g = relu([mean||max] @ W1 + b1) @ W2 + b2 ; one thread per graph.
// ---------------------------------------------------------------------------
__global__ void k_head(const float* __restrict__ psum,
                       const float* __restrict__ pmax,
                       const float* __restrict__ pcnt,
                       const float* __restrict__ W1,
                       const float* __restrict__ b1,
                       const float* __restrict__ W2,
                       const float* __restrict__ b2,
                       float* __restrict__ out) {
  const int g = threadIdx.x;
  if (g >= N_GRAPHS) return;
  float gv[2 * H];
  const float c = fmaxf(pcnt[g], 1.0f);
  for (int j = 0; j < H; ++j) gv[j] = psum[g * H + j] / c;
  for (int j = 0; j < H; ++j) {
    float v = pmax[g * H + j];
    if (!(v > -3.0e38f)) v = 0.0f;  // -inf (empty graph) -> 0
    gv[H + j] = v;
  }
  float y = b2[0];
  for (int o = 0; o < H; ++o) {
    float acc = b1[o];
    for (int j = 0; j < 2 * H; ++j) acc += gv[j] * W1[j * H + o];
    acc = acc > 0.0f ? acc : 0.0f;
    y += acc * W2[o];
  }
  out[g] = y;
}

// ---------------------------------------------------------------------------
extern "C" void kernel_launch(void* const* d_in, const int* in_sizes, int n_in,
                              void* d_out, int out_size, void* d_ws, size_t ws_size,
                              hipStream_t stream) {
  const float* x       = (const float*)d_in[0];
  const int*   eidx    = (const int*)  d_in[1];
  const int*   batch   = (const int*)  d_in[2];
  const float* W_in    = (const float*)d_in[3];
  const float* b_in    = (const float*)d_in[4];
  const float* a_att   = (const float*)d_in[5];
  const float* Wi      = (const float*)d_in[6];
  const float* Wh      = (const float*)d_in[7];
  const float* bi      = (const float*)d_in[8];
  const float* bh      = (const float*)d_in[9];
  const float* W1      = (const float*)d_in[10];
  const float* b1      = (const float*)d_in[11];
  const float* W2      = (const float*)d_in[12];
  const float* b2      = (const float*)d_in[13];
  float* out = (float*)d_out;

  const int* e_src = eidx;
  const int* e_dst = eidx + N_EDGES;

  // Carve workspace (256B aligned slices)
  uintptr_t p = (uintptr_t)d_ws;
  auto carve = [&](size_t bytes) -> float* {
    p = (p + 255) & ~(uintptr_t)255;
    float* r = (float*)p;
    p += bytes;
    return r;
  };
  float* h      = carve((size_t)N_NODES * H * 4);
  float* m      = carve((size_t)N_NODES * H * 4);
  float* s_src  = carve((size_t)N_NODES * 4);
  float* s_dst  = carve((size_t)N_NODES * 4);
  float* segmax = carve((size_t)N_NODES * 4);
  float* segsum = carve((size_t)N_NODES * 4);
  float* score  = carve((size_t)N_EDGES * 4);
  float* psum   = carve((size_t)N_GRAPHS * H * 4);
  float* pmax   = carve((size_t)N_GRAPHS * H * 4);
  float* pcnt   = carve((size_t)N_GRAPHS * 4);
  (void)ws_size; (void)in_sizes; (void)n_in; (void)out_size;

  const int NT16 = N_NODES / 16;  // 3125 node tiles

  // 1. Input projection (WMMA)
  k_input_proj<<<NT16, 128, 0, stream>>>(x, W_in, b_in, h);

  // 2. Message-passing timesteps
  for (int t = 0; t < T_STEPS; ++t) {
    k_init_step<<<(N_NODES * H + 255) / 256, 256, 0, stream>>>(m, segmax, segsum);
    k_att_scalar<<<(N_NODES + 255) / 256, 256, 0, stream>>>(h, a_att, s_src, s_dst);
    k_edge_score<<<(N_EDGES + 255) / 256, 256, 0, stream>>>(e_src, e_dst, s_src, s_dst,
                                                            score, segmax);
    k_edge_exp<<<(N_EDGES + 255) / 256, 256, 0, stream>>>(e_dst, score, segmax, segsum);
    k_edge_msg<<<(N_EDGES * 32 + 255) / 256, 256, 0, stream>>>(e_src, e_dst, score,
                                                               segsum, h, m);
    k_gru<<<(NT16 + 3) / 4, 128, 0, stream>>>(m, h, Wi, Wh, bi, bh);
  }

  // 3. Readout + head
  k_init_pool<<<(N_GRAPHS * H + 255) / 256, 256, 0, stream>>>(psum, pmax, pcnt);
  k_pool<<<(N_NODES * 32 + 255) / 256, 256, 0, stream>>>(h, batch, psum, pmax, pcnt);
  k_head<<<1, 64, 0, stream>>>(psum, pmax, pcnt, W1, b1, W2, b2, out);
}
